// MPNN_29437705846952
// MI455X (gfx1250) — matrix operand order repaired and checked by reading, not compiled, not run on hardware
//
#include <hip/hip_runtime.h>
#include <stdint.h>

typedef __bf16 bf16;
typedef __attribute__((ext_vector_type(16))) __bf16 v16bf;
typedef __attribute__((ext_vector_type(8)))  float  v8f;

#define LRELU_SLOPE 0.8f
#define BN_EPS 1e-5f

// ---------------------------------------------------------------------------
// We tile permutation: logical BN column c = d*64 + o  <->  stored offset
// s = (o/2)*128 + d*2 + (o&1).  Lane L of the message wave then reads a
// contiguous 256-byte slice (outputs o=2L,2L+1 for all d) with b128 loads.
// ---------------------------------------------------------------------------
__device__ __forceinline__ int we_store_of_col(int c) {
  const int d = c >> 6, o = c & 63;
  return ((o >> 1) << 7) | (d << 1) | (o & 1);
}
__device__ __forceinline__ int we_col_of_store(int s) {
  const int d = (s & 127) >> 1;
  const int o = ((s >> 7) << 1) | (s & 1);
  return (d << 6) | o;
}

// ---------------------------------------------------------------------------
// WMMA A fragment (ISA 7.12.2): lane = m (0-15) / m (16-31 = other K-half).
// Two contiguous 8-bf16 runs per lane -> compiler emits 2x ds_load_b128.
// ---------------------------------------------------------------------------
__device__ __forceinline__ v16bf load_a_frag(const bf16* As, int lda, int lane) {
  const int half = lane >> 4;
  const int m    = lane & 15;
  v16bf a;
#pragma unroll
  for (int j = 0; j < 8; ++j) {
    const int kbase = (half ? 16 : 0) + 2 * j;
    a[2 * j]     = As[m * lda + kbase];
    a[2 * j + 1] = As[m * lda + kbase + 1];
  }
  return a;
}

// ---------------------------------------------------------------------------
// Generic WMMA GEMM, K templated (fully unrolled k-loop).
//   C[M x NC] = Aconv(bf16) * Bconv(bf16) + bias
//   A  : fp32 [M x K] row-major (lda == K), optional per-row scale
//   B  : fp32 ; bT==0 -> [K x NC] row-major ; bT==1 -> [NC x K] (uses B^T)
//   out: fp32 (Cf) or bf16 (Cbf); permOut!=0 applies we_store_of_col to cols.
// Block = 128 threads (4 waves); tile = 16 rows x 64 cols; wave = 16x16.
// B is staged in LDS in *fragment order* [chunk][ctile][lane][16bf16] so a
// B fragment is one contiguous 32-byte load per lane (2x ds_load_b128).
// ---------------------------------------------------------------------------
template<int K>
__global__ __launch_bounds__(128)
void gemm_wmma_bf16(const float* __restrict__ A, const float* __restrict__ B,
                    const float* __restrict__ bias, const float* __restrict__ rowScale,
                    float* __restrict__ Cf, bf16* __restrict__ Cbf,
                    int M, int NC, int bT, int permOut)
{
  constexpr int NCH = K / 32;              // k-chunks of 32
  __shared__ bf16 As[16 * K];              // row-major A tile
  __shared__ bf16 Bs[NCH * 4 * 32 * 16];   // fragment-order B tiles
  const int tid  = threadIdx.x;
  const int wave = tid >> 5;
  const int lane = tid & 31;
  const int row0 = blockIdx.x * 16;
  const int col0 = blockIdx.y * 64;

  // Stage A: 8 contiguous elements per unit -> vector global loads + b128 LDS store
  for (int u = tid; u < 16 * (K / 8); u += 128) {
    const int r  = u / (K / 8);
    const int c8 = (u - r * (K / 8)) * 8;
    const float* ap = A + (size_t)(row0 + r) * K + c8;
    const float  sc = rowScale ? rowScale[row0 + r] : 1.f;
#pragma unroll
    for (int t = 0; t < 8; ++t) As[r * K + c8 + t] = (bf16)(ap[t] * sc);
  }
  // Stage B in fragment order: unit = (chunk, ctile, frag-lane) -> 16 contiguous bf16
  for (int u = tid; u < NCH * 128; u += 128) {
    const int chunk = u >> 7;
    const int rest  = u & 127;
    const int lu    = rest & 31;
    const int kbase = chunk * 32 + ((lu >> 4) << 4);
    const int n     = col0 + ((rest >> 5) << 4) + (lu & 15);
    bf16* dp = Bs + u * 16;
#pragma unroll
    for (int t = 0; t < 16; ++t) {
      const int k = kbase + t;
      const float v = bT ? B[(size_t)n * K + k] : B[(size_t)k * NC + n];
      dp[t] = (bf16)v;
    }
  }
  __syncthreads();

  v8f acc = {0.f, 0.f, 0.f, 0.f, 0.f, 0.f, 0.f, 0.f};
#pragma unroll
  for (int ch = 0; ch < NCH; ++ch) {
    const v16bf a = load_a_frag(As + ch * 32, K, lane);
    const v16bf b = *(const v16bf*)(Bs + ((size_t)((ch * 4 + wave) * 32 + lane)) * 16);
    acc = __builtin_amdgcn_wmma_f32_16x16x32_bf16(false, a, false, b,
                                                  (short)0, acc, false, false);
  }

  // C/D layout: VGPR j -> M = j + 8*(lane/16), N = lane%16
  const int half = lane >> 4, n = lane & 15;
  const int col  = col0 + wave * 16 + n;
  const float bv = bias ? bias[col] : 0.f;
  const int  scol = permOut ? we_store_of_col(col) : col;
#pragma unroll
  for (int j = 0; j < 8; ++j) {
    const int row = row0 + half * 8 + j;
    const float v = acc[j] + bv;
    if (Cbf) Cbf[(size_t)row * NC + scol] = (bf16)v;
    else     Cf [(size_t)row * NC + scol] = v;
  }
}

// ---------------------------------------------------------------------------
// Small utility kernels
// ---------------------------------------------------------------------------
__global__ void zero_f32(float* __restrict__ p, int n) {
  const int i = blockIdx.x * 256 + threadIdx.x;
  if (i < n) p[i] = 0.f;
}

__global__ void deg_accum(const int* __restrict__ dst, float* __restrict__ deg, int E) {
  const int e = blockIdx.x * 256 + threadIdx.x;
  if (e < E) atomicAdd(&deg[dst[e]], 1.f);
}

__global__ void inv_deg_k(const float* __restrict__ deg, float* __restrict__ inv, int n) {
  const int i = blockIdx.x * 256 + threadIdx.x;
  if (i < n) inv[i] = 1.f / fmaxf(deg[i], 1.f);
}

// Coalesced column partial sums over stored order (thread = column)
__global__ void col_accum(const void* __restrict__ X, int isbf16, int R, int C,
                          int rowsPerBlock, float* __restrict__ colsum,
                          float* __restrict__ colsq)
{
  const int c = blockIdx.x * blockDim.x + threadIdx.x;
  if (c >= C) return;
  const int r0 = blockIdx.y * rowsPerBlock;
  const int r1 = (r0 + rowsPerBlock < R) ? (r0 + rowsPerBlock) : R;
  float s = 0.f, s2 = 0.f;
  if (isbf16) {
    const bf16* x = (const bf16*)X;
    for (int r = r0; r < r1; ++r) { const float v = (float)x[(size_t)r * C + c]; s += v; s2 += v * v; }
  } else {
    const float* x = (const float*)X;
    for (int r = r0; r < r1; ++r) { const float v = x[(size_t)r * C + c]; s += v; s2 += v * v; }
  }
  atomicAdd(&colsum[c], s);
  atomicAdd(&colsq[c], s2);
}

// mapStore!=0: index c is a *stored* offset; gamma/beta fetched at logical col.
__global__ void bn_finalize(const float* __restrict__ colsum, const float* __restrict__ colsq,
                            const float* __restrict__ g, const float* __restrict__ beta,
                            float* __restrict__ scale, float* __restrict__ shift,
                            int C, float invR, int mapStore)
{
  const int c = blockIdx.x * 256 + threadIdx.x;
  if (c >= C) return;
  const int cl = mapStore ? we_col_of_store(c) : c;
  const float m  = colsum[c] * invR;
  const float v  = colsq[c] * invR - m * m;
  const float sc = g[cl] * rsqrtf(v + BN_EPS);
  scale[c] = sc;
  shift[c] = beta[cl] - m * sc;
}

__global__ void bn_leaky_inplace(float* __restrict__ Y, const float* __restrict__ scale,
                                 const float* __restrict__ shift, int total, int C)
{
  const int i = blockIdx.x * 256 + threadIdx.x;
  if (i >= total) return;
  const int c = i % C;
  float v = Y[i] * scale[c] + shift[c];
  Y[i] = (v >= 0.f) ? v : LRELU_SLOPE * v;
}

__global__ void bn_apply_bf16(bf16* __restrict__ Y, const float* __restrict__ scale,
                              const float* __restrict__ shift, long long total, int C)
{
  const long long i = (long long)blockIdx.x * 256 + threadIdx.x;
  if (i >= total) return;
  const int c = (int)(i % C);
  const float v = (float)Y[i];
  Y[i] = (bf16)(v * scale[c] + shift[c]);
}

// ---------------------------------------------------------------------------
// Per-edge message: msg[e,:] = h[src[e],:] @ We[e] (bf16, lane-tiled layout),
// scatter-added into agg[dst[e],:]. One wave per edge; lane owns outputs
// o = 2*lane, 2*lane+1 and streams its contiguous 256 B slice of We[e] with
// 16x global_load_b128 (1 FLOP/byte -> pure HBM stream, VALU FMA suffices).
// ---------------------------------------------------------------------------
__global__ __launch_bounds__(256)
void edge_message(const float* __restrict__ h, const bf16* __restrict__ We,
                  const int* __restrict__ src, const int* __restrict__ dst,
                  float* __restrict__ agg, int E)
{
  __shared__ float hs[8][64];
  const int wave = threadIdx.x >> 5;
  const int lane = threadIdx.x & 31;
  const int e    = blockIdx.x * 8 + wave;
  const bool valid = (e < E);
  int s = 0, d = 0;
  if (valid) {
    s = src[e]; d = dst[e];
    hs[wave][2 * lane]     = h[(size_t)s * 64 + 2 * lane];
    hs[wave][2 * lane + 1] = h[(size_t)s * 64 + 2 * lane + 1];
  }
  __syncthreads();
  if (!valid) return;

  if (e + 8 < E) __builtin_prefetch(We + (size_t)(e + 8) * 4096 + lane * 128, 0, 1);

  const uint4* __restrict__ w4 =
      (const uint4*)(We + (size_t)e * 4096 + (size_t)lane * 128);
  float a0 = 0.f, a1 = 0.f;
#pragma unroll
  for (int t = 0; t < 16; ++t) {
    const uint4 p = w4[t];                 // d = 4t .. 4t+3, both outputs
    const float* hv = &hs[wave][4 * t];
    a0 = fmaf(hv[0], __uint_as_float(p.x << 16),         a0);
    a1 = fmaf(hv[0], __uint_as_float(p.x & 0xffff0000u), a1);
    a0 = fmaf(hv[1], __uint_as_float(p.y << 16),         a0);
    a1 = fmaf(hv[1], __uint_as_float(p.y & 0xffff0000u), a1);
    a0 = fmaf(hv[2], __uint_as_float(p.z << 16),         a0);
    a1 = fmaf(hv[2], __uint_as_float(p.z & 0xffff0000u), a1);
    a0 = fmaf(hv[3], __uint_as_float(p.w << 16),         a0);
    a1 = fmaf(hv[3], __uint_as_float(p.w & 0xffff0000u), a1);
  }
  atomicAdd(&agg[(size_t)d * 64 + 2 * lane],     a0);
  atomicAdd(&agg[(size_t)d * 64 + 2 * lane + 1], a1);
}

// ---------------------------------------------------------------------------
// GRU gates (PyTorch r,z,n order): h' = (1-z)*tanh(in + r*hn) + z*h
// ---------------------------------------------------------------------------
__global__ void gru_elem(const float* __restrict__ gi, const float* __restrict__ gh,
                         const float* __restrict__ hp, float* __restrict__ ho, int total)
{
  const int i = blockIdx.x * 256 + threadIdx.x;
  if (i >= total) return;
  const int node = i >> 6, j = i & 63;
  const float* a = gi + (size_t)node * 192;
  const float* b = gh + (size_t)node * 192;
  const float r  = 1.f / (1.f + __expf(-(a[j]       + b[j])));
  const float z  = 1.f / (1.f + __expf(-(a[64 + j]  + b[64 + j])));
  const float nn = tanhf(a[128 + j] + r * b[128 + j]);
  ho[i] = (1.f - z) * nn + z * hp[i];
}

// ---------------------------------------------------------------------------
// Host orchestration
// ---------------------------------------------------------------------------
extern "C" void kernel_launch(void* const* d_in, const int* in_sizes, int n_in,
                              void* d_out, int out_size, void* d_ws, size_t ws_size,
                              hipStream_t stream)
{
  const float* h_node = (const float*)d_in[0];
  const float* h_edge = (const float*)d_in[1];
  const int*   src    = (const int*)  d_in[2];
  const int*   dst    = (const int*)  d_in[3];
  const float* W1     = (const float*)d_in[4];
  const float* b1     = (const float*)d_in[5];
  const float* g1     = (const float*)d_in[6];
  const float* beta1  = (const float*)d_in[7];
  const float* W2     = (const float*)d_in[8];
  const float* b2     = (const float*)d_in[9];
  const float* g2     = (const float*)d_in[10];
  const float* beta2  = (const float*)d_in[11];
  const float* Wi     = (const float*)d_in[12];
  const float* Wh     = (const float*)d_in[13];
  const float* bi     = (const float*)d_in[14];
  const float* bh     = (const float*)d_in[15];
  (void)n_in; (void)out_size; (void)ws_size;

  const int E  = in_sizes[2];          // 60000
  const int Nn = in_sizes[0] / 64;     // 12000
  const int D = 64, DE = 128, DD = 4096, G3 = 192;

  // Bump-allocate workspace (~560 MB)
  char* wsb = (char*)d_ws; size_t off = 0;
  auto alloc = [&](size_t bytes) -> char* {
    char* p = wsb + off; off = (off + bytes + 255) & ~(size_t)255; return p;
  };
  float* Y1    = (float*)alloc((size_t)E * DE * sizeof(float));
  bf16*  We    = (bf16*) alloc((size_t)E * DD * sizeof(bf16));   // 492 MB, lane-tiled
  float* agg   = (float*)alloc((size_t)Nn * D  * sizeof(float));
  float* gi    = (float*)alloc((size_t)Nn * G3 * sizeof(float));
  float* gh    = (float*)alloc((size_t)Nn * G3 * sizeof(float));
  float* hb0   = (float*)alloc((size_t)Nn * D  * sizeof(float));
  float* hb1   = (float*)alloc((size_t)Nn * D  * sizeof(float));
  float* deg   = (float*)alloc((size_t)Nn * sizeof(float));
  float* invd  = (float*)alloc((size_t)Nn * sizeof(float));
  float* csum  = (float*)alloc(4096 * sizeof(float));
  float* csq   = (float*)alloc(4096 * sizeof(float));
  float* scale = (float*)alloc(4096 * sizeof(float));
  float* shift = (float*)alloc(4096 * sizeof(float));

  // ---- degrees / inverse degrees --------------------------------------
  zero_f32 <<<(Nn + 255) / 256, 256, 0, stream>>>(deg, Nn);
  deg_accum<<<(E  + 255) / 256, 256, 0, stream>>>(dst, deg, E);
  inv_deg_k<<<(Nn + 255) / 256, 256, 0, stream>>>(deg, invd, Nn);

  // ---- Edge network (loop-invariant: computed ONCE, reused 3 layers) --
  gemm_wmma_bf16<128><<<dim3(E / 16, DE / 64), 128, 0, stream>>>(
      h_edge, W1, b1, nullptr, Y1, nullptr, E, DE, 0, 0);

  zero_f32<<<1, 256, 0, stream>>>(csum, DE);
  zero_f32<<<1, 256, 0, stream>>>(csq,  DE);
  col_accum<<<dim3(1, 64), 256, 0, stream>>>(Y1, 0, E, DE, (E + 63) / 64, csum, csq);
  bn_finalize<<<1, 256, 0, stream>>>(csum, csq, g1, beta1, scale, shift, DE,
                                     1.f / (float)E, 0);
  bn_leaky_inplace<<<(E * DE + 255) / 256, 256, 0, stream>>>(Y1, scale, shift, E * DE, DE);

  // GEMM2 writes We in the lane-tiled layout (permOut=1)
  gemm_wmma_bf16<128><<<dim3(E / 16, DD / 64), 128, 0, stream>>>(
      Y1, W2, b2, nullptr, nullptr, We, E, DD, 0, 1);

  zero_f32<<<(DD + 255) / 256, 256, 0, stream>>>(csum, DD);
  zero_f32<<<(DD + 255) / 256, 256, 0, stream>>>(csq,  DD);
  col_accum<<<dim3(DD / 256, 64), 256, 0, stream>>>(We, 1, E, DD, (E + 63) / 64, csum, csq);
  bn_finalize<<<(DD + 255) / 256, 256, 0, stream>>>(csum, csq, g2, beta2, scale, shift, DD,
                                                    1.f / (float)E, 1);
  {
    const long long tot = (long long)E * DD;
    bn_apply_bf16<<<(unsigned)((tot + 255) / 256), 256, 0, stream>>>(We, scale, shift, tot, DD);
  }

  // ---- 3 message-passing layers ---------------------------------------
  const float* hcur = h_node;
  for (int l = 0; l < 3; ++l) {
    float* hnext = (l == 0) ? hb0 : (l == 1) ? hb1 : (float*)d_out;

    zero_f32<<<(Nn * D + 255) / 256, 256, 0, stream>>>(agg, Nn * D);
    edge_message<<<(E + 7) / 8, 256, 0, stream>>>(hcur, We, src, dst, agg, E);

    // gi = (agg * inv_deg) @ Wi^T + bi   (scatter-mean folded in via rowScale)
    gemm_wmma_bf16<64><<<dim3(Nn / 16, G3 / 64), 128, 0, stream>>>(
        agg, Wi, bi, invd, gi, nullptr, Nn, G3, 1, 0);
    // gh = hcur @ Wh^T + bh
    gemm_wmma_bf16<64><<<dim3(Nn / 16, G3 / 64), 128, 0, stream>>>(
        hcur, Wh, bh, nullptr, gh, nullptr, Nn, G3, 1, 0);

    gru_elem<<<(Nn * D + 255) / 256, 256, 0, stream>>>(gi, gh, hcur, hnext, Nn * D);
    hcur = hnext;
  }
}
